// PCImageAligner_70171175682074
// MI455X (gfx1250) — matrix operand order repaired and checked by reading, not compiled, not run on hardware
//
#include <hip/hip_runtime.h>
#include <hip/hip_bf16.h>

typedef __attribute__((ext_vector_type(16))) _Float16 v16h;
typedef __attribute__((ext_vector_type(8)))  float    v8f;

#define BDIM 256
#define BM 128
#define BN 64
#define BK 32
#define LDA 40   // padded LDS stride (halves) for A tile  [128][40]
#define LDB 40   // padded LDS stride (halves) for B tile (transposed [n][k]) [64][40]

static constexpr int Bb = 4, NP = 8192, NI = 1024, DP = 512, DI = 768, HID = 512, DO = 512;

// ---------------- fp32 -> fp16 conversion (grid-stride) ----------------
__global__ __launch_bounds__(BDIM) void cvt_f32_f16(const float* __restrict__ in,
                                                    _Float16* __restrict__ out, int n) {
  for (int i = blockIdx.x * BDIM + threadIdx.x; i < n; i += gridDim.x * BDIM)
    out[i] = (_Float16)in[i];
}

// ---------------- WMMA GEMM: C = act(A[MxK] * W[KxN] + bias) ----------------
// MODE 0: relu -> f16   MODE 1: plain -> f16   MODE 2: sigmoid -> f16
// MODE 3: outF = resid + (float)gateH * (acc + bias)   (final fusion, f32 out)
//
// Tiling: 128x64 block tile, 8 wave32s; wave w owns a 16(M) x 64(N) slab ->
// 4 accumulators, 4 v_wmma_f32_16x16x32_f16 per k-step, A fragment loaded once.
// Register-stage pipeline: next k-step's global tiles are loaded while the
// current step's WMMAs run.
//
// Fragment layouts per CDNA5 ISA 7.12.2 (16-bit, wave32):
//  A 16x32: lanes 0-15: M=lane, halves = K[0..7],K[16..23]; lanes 16-31: K[8..15],K[24..31]
//  B 32x16: lanes 0-15: N=lane, halves = K[0..7],K[16..23]; lanes 16-31: K[8..15],K[24..31]
// W tile is stored transposed in LDS so both fragments are contiguous 8-half groups.
template <int MODE>
__global__ __launch_bounds__(BDIM) void gemm_wmma_f16(
    const _Float16* __restrict__ A, const _Float16* __restrict__ W,
    const float* __restrict__ bias,
    _Float16* __restrict__ outH, float* __restrict__ outF,
    const _Float16* __restrict__ gateH, const float* __restrict__ resid,
    int M, int N, int K) {
  __shared__ alignas(16) _Float16 sA[BM * LDA];
  __shared__ alignas(16) _Float16 sB[BN * LDB];

  const int tid  = threadIdx.x;
  const int lane = tid & 31;
  const int wave = tid >> 5;          // wave = M-tile index (0..7)
  const int mBase = blockIdx.y * BM;
  const int nBase = blockIdx.x * BN;

  const int aRow = tid >> 1;          // 0..127
  const int aCol = (tid & 1) << 4;    // 0 or 16 (halves)
  const int bRow = tid >> 3;          // 0..31 = k within tile
  const int bSeg = (tid & 7) << 3;    // 0..56 = n within tile

  const int mrow = wave * 16 + (lane & 15);
  const int kbA  = (lane & 16) ? 8 : 0;
  const int kbB  = (lane & 16) ? 8 : 0;
  const int ncol = lane & 15;

  v8f acc[4] = {v8f{}, v8f{}, v8f{}, v8f{}};

  const _Float16* aPtr = A + (size_t)(mBase + aRow) * K + aCol;
  const _Float16* wPtr = W + (size_t)bRow * N + nBase + bSeg;
  const size_t wStep = (size_t)BK * N;

  uint4 a0, a1;
  union { uint4 q; _Float16 h[8]; } wv;
  // stage k-step 0 into registers
  a0 = *(const uint4*)(aPtr);
  a1 = *(const uint4*)(aPtr + 8);
  wv.q = *(const uint4*)(wPtr);

  const int ksteps = K / BK;
  for (int kk = 0; kk < ksteps; ++kk) {
    __syncthreads();                         // prior compute done reading LDS
    *(uint4*)(&sA[aRow * LDA + aCol])     = a0;
    *(uint4*)(&sA[aRow * LDA + aCol + 8]) = a1;
#pragma unroll
    for (int j = 0; j < 8; ++j) sB[(bSeg + j) * LDB + bRow] = wv.h[j];  // transpose
    __syncthreads();

    if (kk + 1 < ksteps) {                   // overlap next global loads w/ WMMAs
      const _Float16* aN = aPtr + (kk + 1) * BK;
      const _Float16* wN = wPtr + (kk + 1) * wStep;
      a0 = *(const uint4*)(aN);
      a1 = *(const uint4*)(aN + 8);
      wv.q = *(const uint4*)(wN);
      if (kk + 2 < ksteps) {                 // global_prefetch_b8, near-cache scope
        __builtin_prefetch(aPtr + (kk + 2) * BK, 0, 3);
        __builtin_prefetch(wPtr + (kk + 2) * wStep, 0, 3);
      }
    }

    union { uint4 q[2]; v16h v; } fa;
    fa.q[0] = *(const uint4*)(&sA[mrow * LDA + kbA]);
    fa.q[1] = *(const uint4*)(&sA[mrow * LDA + kbA + 16]);
#pragma unroll
    for (int nt = 0; nt < 4; ++nt) {
      union { uint4 q[2]; v16h v; } fb;
      const int nc = nt * 16 + ncol;
      fb.q[0] = *(const uint4*)(&sB[nc * LDB + kbB]);
      fb.q[1] = *(const uint4*)(&sB[nc * LDB + kbB + 16]);
      acc[nt] = __builtin_amdgcn_wmma_f32_16x16x32_f16(false, fa.v, false, fb.v,
                                                       (short)0, acc[nt], false, false);
    }
  }

  // Epilogue. C layout: lane 0-15 -> N=lane, VGPR r -> M=r; lanes 16-31 -> M=r+8.
  const int mlOff = (lane & 16) ? 8 : 0;
#pragma unroll
  for (int nt = 0; nt < 4; ++nt) {
    const int col = nBase + nt * 16 + ncol;
    const float bv = bias[col];
#pragma unroll
    for (int r = 0; r < 8; ++r) {
      const int row = mBase + wave * 16 + mlOff + r;
      const float v = acc[nt][r] + bv;
      const size_t off = (size_t)row * N + col;
      if (MODE == 0)      outH[off] = (_Float16)fmaxf(v, 0.f);
      else if (MODE == 1) outH[off] = (_Float16)v;
      else if (MODE == 2) outH[off] = (_Float16)(1.f / (1.f + __expf(-v)));
      else                outF[off] = resid[off] + (float)gateH[off] * v;
    }
  }
}

// ---------------- masked 3-NN + inverse-distance weights ----------------
__global__ __launch_bounds__(BDIM) void knn_top3(const float* __restrict__ centers,
                                                 const float* __restrict__ coords,
                                                 const int* __restrict__ mask,
                                                 int* __restrict__ outIdx,
                                                 float* __restrict__ outW) {
  __shared__ float sx[NI], sy[NI], sz[NI], sp[NI];
  const int b = blockIdx.y;
  const int p = blockIdx.x * BDIM + threadIdx.x;
  for (int i = threadIdx.x; i < NI; i += BDIM) {
    const float* c = coords + ((size_t)b * NI + i) * 3;
    sx[i] = c[0]; sy[i] = c[1]; sz[i] = c[2];
    sp[i] = mask[b * NI + i] ? 0.f : 3.0e38f;  // invalid -> +inf-ish
  }
  __syncthreads();
  const float* q = centers + ((size_t)b * NP + p) * 3;
  const float qx = q[0], qy = q[1], qz = q[2];
  float d0 = 3.3e38f, d1 = 3.3e38f, d2 = 3.3e38f;
  int   i0 = 0, i1 = 0, i2 = 0;
  for (int i = 0; i < NI; ++i) {
    const float dx = qx - sx[i], dy = qy - sy[i], dz = qz - sz[i];
    const float d = fmaxf(dx * dx + dy * dy + dz * dz, 0.f) + sp[i];
    if (d < d2) {
      if (d < d0)      { d2 = d1; i2 = i1; d1 = d0; i1 = i0; d0 = d; i0 = i; }
      else if (d < d1) { d2 = d1; i2 = i1; d1 = d;  i1 = i; }
      else             { d2 = d;  i2 = i; }
    }
  }
  const float w0 = (d0 < 1e38f) ? 1.f / fmaxf(sqrtf(d0), 1e-6f) : 0.f;
  const float w1 = (d1 < 1e38f) ? 1.f / fmaxf(sqrtf(d1), 1e-6f) : 0.f;
  const float w2 = (d2 < 1e38f) ? 1.f / fmaxf(sqrtf(d2), 1e-6f) : 0.f;
  const float s  = fmaxf(w0 + w1 + w2, 1e-6f);
  const size_t o = ((size_t)b * NP + p) * 3;
  outIdx[o] = i0; outIdx[o + 1] = i1; outIdx[o + 2] = i2;
  outW[o] = w0 / s; outW[o + 1] = w1 / s; outW[o + 2] = w2 / s;
}

// ---------------- gather + weighted sum + concat -> f16 xcat ----------------
__global__ __launch_bounds__(BDIM) void build_xcat(const float* __restrict__ point,
                                                   const _Float16* __restrict__ feat,
                                                   const int* __restrict__ knnIdx,
                                                   const float* __restrict__ knnW,
                                                   _Float16* __restrict__ xcat) {
  const int row = blockIdx.x;      // 0 .. B*Np-1
  const int b   = row >> 13;       // Np = 8192
  const size_t o3 = (size_t)row * 3;
  const int   i0 = knnIdx[o3], i1 = knnIdx[o3 + 1], i2 = knnIdx[o3 + 2];
  const float w0 = knnW[o3],  w1 = knnW[o3 + 1],  w2 = knnW[o3 + 2];
  const _Float16* f0 = feat + ((size_t)b * NI + i0) * DO;
  const _Float16* f1 = feat + ((size_t)b * NI + i1) * DO;
  const _Float16* f2 = feat + ((size_t)b * NI + i2) * DO;
  const float* pt = point + (size_t)row * DP;
  _Float16* xo = xcat + (size_t)row * (DP + DO);
  for (int c = threadIdx.x; c < DP; c += BDIM) xo[c] = (_Float16)pt[c];
  for (int c = threadIdx.x; c < DO; c += BDIM)
    xo[DP + c] = (_Float16)(w0 * (float)f0[c] + w1 * (float)f1[c] + w2 * (float)f2[c]);
}

extern "C" void kernel_launch(void* const* d_in, const int* in_sizes, int n_in,
                              void* d_out, int out_size, void* d_ws, size_t ws_size,
                              hipStream_t stream) {
  (void)in_sizes; (void)n_in; (void)out_size; (void)ws_size;
  const float* point   = (const float*)d_in[0];
  const float* centers = (const float*)d_in[1];
  const float* imgTok  = (const float*)d_in[2];
  const float* coords  = (const float*)d_in[3];
  const int*   maskI   = (const int*)d_in[4];
  const float* img_w1   = (const float*)d_in[5],  *img_b1   = (const float*)d_in[6];
  const float* img_w2   = (const float*)d_in[7],  *img_b2   = (const float*)d_in[8];
  const float* gate_w1  = (const float*)d_in[9],  *gate_b1  = (const float*)d_in[10];
  const float* gate_w2  = (const float*)d_in[11], *gate_b2  = (const float*)d_in[12];
  const float* delta_w1 = (const float*)d_in[13], *delta_b1 = (const float*)d_in[14];
  const float* delta_w2 = (const float*)d_in[15], *delta_b2 = (const float*)d_in[16];

  char* ws = (char*)d_ws;
  size_t off = 0;
  auto alloc = [&](size_t bytes) -> void* {
    void* p = ws + off;
    off += (bytes + 255) & ~(size_t)255;
    return p;
  };
  _Float16* wImg1  = (_Float16*)alloc((size_t)DI * HID * 2);
  _Float16* wImg2  = (_Float16*)alloc((size_t)HID * DO * 2);
  _Float16* wGate1 = (_Float16*)alloc((size_t)(DP + DO) * HID * 2);
  _Float16* wGate2 = (_Float16*)alloc((size_t)HID * DO * 2);
  _Float16* wDel1  = (_Float16*)alloc((size_t)(DP + DO) * HID * 2);
  _Float16* wDel2  = (_Float16*)alloc((size_t)HID * DO * 2);
  _Float16* xiH    = (_Float16*)alloc((size_t)Bb * NI * DI * 2);
  _Float16* hImg   = (_Float16*)alloc((size_t)Bb * NI * HID * 2);
  _Float16* featH  = (_Float16*)alloc((size_t)Bb * NI * DO * 2);
  int*      knnIdx = (int*)alloc((size_t)Bb * NP * 3 * 4);
  float*    knnW   = (float*)alloc((size_t)Bb * NP * 3 * 4);
  _Float16* xcat   = (_Float16*)alloc((size_t)Bb * NP * (DP + DO) * 2);
  _Float16* hG     = (_Float16*)alloc((size_t)Bb * NP * HID * 2);
  _Float16* hD     = (_Float16*)alloc((size_t)Bb * NP * HID * 2);
  _Float16* gateH  = (_Float16*)alloc((size_t)Bb * NP * DO * 2);

  auto cvt = [&](const float* src, _Float16* dst, int n) {
    cvt_f32_f16<<<dim3((n + BDIM - 1) / BDIM), BDIM, 0, stream>>>(src, dst, n);
  };
  cvt(img_w1,   wImg1,  DI * HID);
  cvt(img_w2,   wImg2,  HID * DO);
  cvt(gate_w1,  wGate1, (DP + DO) * HID);
  cvt(gate_w2,  wGate2, HID * DO);
  cvt(delta_w1, wDel1,  (DP + DO) * HID);
  cvt(delta_w2, wDel2,  HID * DO);
  cvt(imgTok,   xiH,    Bb * NI * DI);

  const int Mimg = Bb * NI;      // 4096
  const int Mpt  = Bb * NP;      // 32768
  dim3 gImg(HID / BN, Mimg / BM);  // (8, 32)
  dim3 gPt(HID / BN,  Mpt / BM);   // (8, 256)

  // image MLP: relu(Xi*W1+b1) -> hImg ; hImg*W2+b2 -> featH (f16)
  gemm_wmma_f16<0><<<gImg, BDIM, 0, stream>>>(xiH, wImg1, img_b1, hImg, nullptr,
                                              nullptr, nullptr, Mimg, HID, DI);
  gemm_wmma_f16<1><<<gImg, BDIM, 0, stream>>>(hImg, wImg2, img_b2, featH, nullptr,
                                              nullptr, nullptr, Mimg, DO, HID);

  // masked 3-NN + weights, then xcat = [point | aligned] in f16
  knn_top3<<<dim3(NP / BDIM, Bb), BDIM, 0, stream>>>(centers, coords, maskI, knnIdx, knnW);
  build_xcat<<<dim3(Bb * NP), BDIM, 0, stream>>>(point, featH, knnIdx, knnW, xcat);

  // fusion MLPs
  gemm_wmma_f16<0><<<gPt, BDIM, 0, stream>>>(xcat, wGate1, gate_b1, hG, nullptr,
                                             nullptr, nullptr, Mpt, HID, DP + DO);
  gemm_wmma_f16<0><<<gPt, BDIM, 0, stream>>>(xcat, wDel1, delta_b1, hD, nullptr,
                                             nullptr, nullptr, Mpt, HID, DP + DO);
  gemm_wmma_f16<2><<<gPt, BDIM, 0, stream>>>(hG, wGate2, gate_b2, gateH, nullptr,
                                             nullptr, nullptr, Mpt, DO, HID);
  // final: d_out = point + gate * (hD*W2+b2)
  gemm_wmma_f16<3><<<gPt, BDIM, 0, stream>>>(hD, wDel2, delta_b2, nullptr, (float*)d_out,
                                             gateH, point, Mpt, DO, HID);
}
